// RefWindowAttention_21474836480491
// MI455X (gfx1250) — matrix operand order repaired and checked by reading, not compiled, not run on hardware
//
#include <hip/hip_runtime.h>
#include <hip/hip_bf16.h>

#define DIMC   192
#define HEADS  6
#define HD     32
#define NTOK   64
#define XS     200      // row stride (bf16 elems) for token-major LDS buffers [64][200]
#define VTS    72       // row stride for V-transposed buffer [192][72]
#define PS     72       // row stride for per-wave P scratch [16][72]
#define BUFE   13824    // elems per big LDS buffer (max(64*200, 192*72))
#define WFRAGS 72       // 12 n-tiles * 6 k-tiles per weight
#define WELEMS (WFRAGS*512)  // bf16 elems per packed weight
#define SCALEQ 0.17677669529663687f  // 32^-0.5

typedef __bf16 bf16;
typedef __attribute__((ext_vector_type(16))) bf16  bf16x16;
typedef __attribute__((ext_vector_type(8)))  bf16  bf16x8;
typedef __attribute__((ext_vector_type(4)))  bf16  bf16x4;
typedef __attribute__((ext_vector_type(8)))  float f32x8;
typedef __attribute__((ext_vector_type(4)))  float f32x4;

static __device__ __forceinline__ f32x8 wmma_bf16(bf16x16 a, bf16x16 b, f32x8 c) {
  return __builtin_amdgcn_wmma_f32_16x16x32_bf16(false, a, false, b, (short)0, c, false, false);
}

static __device__ __forceinline__ bf16x16 cat16(bf16x8 lo, bf16x8 hi) {
  return __builtin_shufflevector(lo, hi, 0,1,2,3,4,5,6,7,8,9,10,11,12,13,14,15);
}

// A-fragment (16xK=32 bf16) from row-major LDS: row = rowBase + (lane&15),
// halfs 0..15 <-> K = 16*(lane>=16) + 0..15 (contiguous)
static __device__ __forceinline__ bf16x16 ldsA(const bf16* base, int rowBase, int colBase,
                                               int stride, int lane) {
  const bf16* r = base + (rowBase + (lane & 15)) * stride + colBase + ((lane >> 4) << 4);
  bf16x8 lo = *(const bf16x8*)(r);
  bf16x8 hi = *(const bf16x8*)(r + 8);
  return cat16(lo, hi);
}

// B-fragment (K=32 x 16) where LDS source is N-major: element (k,n) = base[(rowBase+n)*stride + colBase + k]
// col = lane&15; halfs 0..15 <-> K = 16*(lane>=16) + 0..15 (contiguous)
static __device__ __forceinline__ bf16x16 ldsBT(const bf16* base, int rowBase, int colBase,
                                                int stride, int lane) {
  const bf16* r = base + (rowBase + (lane & 15)) * stride + colBase + ((lane >> 4) << 4);
  bf16x8 lo = *(const bf16x8*)(r);
  bf16x8 hi = *(const bf16x8*)(r + 8);
  return cat16(lo, hi);
}

// pre-packed weight B-fragment: 32 bytes per lane, contiguous
static __device__ __forceinline__ bf16x16 wfrag(const bf16* wbase, int frag, int lane) {
  return *(const bf16x16*)(wbase + frag * 512 + lane * 16);
}

// ---------------- kernel 0: pack f32 weights into WMMA B-fragment bf16 layout ----------------
__global__ void pack_weights(const float* __restrict__ Wq, const float* __restrict__ Wk,
                             const float* __restrict__ Wv, const float* __restrict__ Wp,
                             bf16* __restrict__ out) {
  int gw   = (blockIdx.x * blockDim.x + threadIdx.x) >> 5;
  int lane = threadIdx.x & 31;
  if (gw >= 4 * WFRAGS) return;
  int wsel = gw / WFRAGS, frag = gw % WFRAGS;
  const float* W = (wsel == 0) ? Wq : (wsel == 1) ? Wk : (wsel == 2) ? Wv : Wp;
  int nt  = frag / 6, kt = frag % 6;
  int col = nt * 16 + (lane & 15);
  int k0  = kt * 32 + ((lane >> 4) << 4);
  bf16x16 v;
#pragma unroll
  for (int i = 0; i < 16; ++i) v[i] = (bf16)W[(k0 + i) * DIMC + col];
  *(bf16x16*)(out + wsel * WELEMS + frag * 512 + lane * 16) = v;
}

// GEMM [64,192] @ packed-W[192,192] + bias, optional scale.
// Each wave owns one 16-row block (mt = wave>>1) and 6 consecutive n-tiles
// (nbase = (wave&1)*6): A-fragments hoisted to registers, 6 independent
// accumulators per k-step for XDL pipelining.
static __device__ __forceinline__ void gemm192(const bf16* X, const bf16* W, const float* bias,
                                               float scale, bf16* dst, bool toVt,
                                               int wave, int lane) {
  const int mt    = wave >> 1;
  const int nbase = (wave & 1) * 6;
  bf16x16 A[6];
#pragma unroll
  for (int kt = 0; kt < 6; ++kt) A[kt] = ldsA(X, mt * 16, kt * 32, XS, lane);
  f32x8 acc[6];
#pragma unroll
  for (int i = 0; i < 6; ++i) acc[i] = (f32x8){0.f,0.f,0.f,0.f,0.f,0.f,0.f,0.f};
#pragma unroll
  for (int kt = 0; kt < 6; ++kt) {
#pragma unroll
    for (int i = 0; i < 6; ++i) {
      bf16x16 bb = wfrag(W, (nbase + i) * 6 + kt, lane);
      acc[i] = wmma_bf16(A[kt], bb, acc[i]);
    }
  }
  const int r0 = mt * 16 + ((lane >> 4) << 3);
#pragma unroll
  for (int i = 0; i < 6; ++i) {
    int   col = (nbase + i) * 16 + (lane & 15);
    float bsv = bias[col];
    if (toVt) {
      bf16x8 pk;
#pragma unroll
      for (int v = 0; v < 8; ++v) pk[v] = (bf16)((acc[i][v] + bsv) * scale);
      *(bf16x8*)(dst + col * VTS + r0) = pk;   // contiguous 8 tokens
    } else {
#pragma unroll
      for (int v = 0; v < 8; ++v) dst[(r0 + v) * XS + col] = (bf16)((acc[i][v] + bsv) * scale);
    }
  }
}

// ---------------- main fused kernel: one block = one window ----------------
__global__ __launch_bounds__(256)
void window_attn(const float* __restrict__ xq, const float* __restrict__ xk,
                 const float* __restrict__ xv, const float* __restrict__ mask,
                 const float* __restrict__ bq, const float* __restrict__ bk,
                 const float* __restrict__ bv, const float* __restrict__ bp,
                 const float* __restrict__ bias_table, const bf16* __restrict__ wpack,
                 float* __restrict__ out) {
  __shared__ alignas(32) bf16 smem[4 * BUFE + 8 * 16 * PS];
  bf16* B0 = smem;
  bf16* B1 = smem + BUFE;
  bf16* B2 = smem + 2 * BUFE;
  bf16* B3 = smem + 3 * BUFE;
  const int tid  = threadIdx.x;
  const int wave = tid >> 5, lane = tid & 31;
  bf16* Pw = smem + 4 * BUFE + wave * (16 * PS);
  const int b = blockIdx.x;

  // ---- stage the three inputs f32 -> bf16 into LDS (padded rows) ----
  {
    const f32x4* srcs[3] = { (const f32x4*)(xq + (size_t)b * NTOK * DIMC),
                             (const f32x4*)(xk + (size_t)b * NTOK * DIMC),
                             (const f32x4*)(xv + (size_t)b * NTOK * DIMC) };
    bf16* dsts[3] = { B0, B1, B2 };
#pragma unroll
    for (int t = 0; t < 3; ++t) {
      const f32x4* s = srcs[t];
      bf16* d = dsts[t];
      for (int i = tid; i < (NTOK * DIMC / 4); i += 256) {
        f32x4 val = __builtin_nontemporal_load(s + i);
        int row = i / 48, c = (i % 48) * 4;
        bf16x4 pk;
        pk[0] = (bf16)val.x; pk[1] = (bf16)val.y; pk[2] = (bf16)val.z; pk[3] = (bf16)val.w;
        *(bf16x4*)(d + row * XS + c) = pk;
      }
    }
  }
  __syncthreads();

  // ---- projections: Q (pre-scaled) -> B3, K -> B0, V -> transposed in B1 ----
  gemm192(B0, wpack + 0 * WELEMS, bq, SCALEQ, B3, false, wave, lane);
  __syncthreads();
  gemm192(B1, wpack + 1 * WELEMS, bk, 1.0f, B0, false, wave, lane);
  __syncthreads();
  gemm192(B2, wpack + 2 * WELEMS, bv, 1.0f, B1, true, wave, lane);
  __syncthreads();

  // ---- attention: 24 units (6 heads x 4 row-blocks) over 8 waves ----
  const float* maskw = mask + (size_t)(b & 63) * (NTOK * NTOK);
#pragma unroll
  for (int uu = 0; uu < 3; ++uu) {
    int u  = wave * 3 + uu;
    int h  = u >> 2, mt = u & 3;
    bf16x16 aq = ldsA(B3, mt * 16, h * HD, XS, lane);  // Q rows, one head
    f32x8 s[4];
#pragma unroll
    for (int ct = 0; ct < 4; ++ct) {
      bf16x16 bkf = ldsBT(B0, ct * 16, h * HD, XS, lane);  // K^T fragment
      f32x8 z = {0.f,0.f,0.f,0.f,0.f,0.f,0.f,0.f};
      s[ct] = wmma_bf16(aq, bkf, z);
    }
    const int m  = lane & 15;
    const int lg = (lane >> 4) << 3;
    // relative-position bias + window mask
#pragma unroll
    for (int ct = 0; ct < 4; ++ct) {
      int mm = ct * 16 + m;
#pragma unroll
      for (int v = 0; v < 8; ++v) {
        int n  = mt * 16 + lg + v;
        int dh = (n >> 3) - (mm >> 3) + 7;
        int dw = (n & 7) - (mm & 7) + 7;
        s[ct][v] += bias_table[(dh * 15 + dw) * HEADS + h] + maskw[n * NTOK + mm];
      }
    }
    // row softmax: reduce across 16-lane half-groups (rows identical within a group)
#pragma unroll
    for (int v = 0; v < 8; ++v) {
      float mx = fmaxf(fmaxf(s[0][v], s[1][v]), fmaxf(s[2][v], s[3][v]));
#pragma unroll
      for (int o = 1; o < 16; o <<= 1) mx = fmaxf(mx, __shfl_xor(mx, o, 32));
      float sum = 0.f;
#pragma unroll
      for (int ct = 0; ct < 4; ++ct) { float e = __expf(s[ct][v] - mx); s[ct][v] = e; sum += e; }
#pragma unroll
      for (int o = 1; o < 16; o <<= 1) sum += __shfl_xor(sum, o, 32);
      float inv = 1.0f / sum;
#pragma unroll
      for (int ct = 0; ct < 4; ++ct) s[ct][v] *= inv;
    }
    // P -> per-wave LDS scratch (reshape D-layout -> A-layout), bf16
#pragma unroll
    for (int ct = 0; ct < 4; ++ct)
#pragma unroll
      for (int v = 0; v < 8; ++v)
        Pw[(lg + v) * PS + ct * 16 + m] = (bf16)s[ct][v];

    bf16x16 ap0 = ldsA(Pw, 0, 0, PS, lane);
    bf16x16 ap1 = ldsA(Pw, 0, 32, PS, lane);
#pragma unroll
    for (int dt = 0; dt < 2; ++dt) {
      f32x8 o = {0.f,0.f,0.f,0.f,0.f,0.f,0.f,0.f};
      o = wmma_bf16(ap0, ldsBT(B1, h * HD + dt * 16, 0,  VTS, lane), o);
      o = wmma_bf16(ap1, ldsBT(B1, h * HD + dt * 16, 32, VTS, lane), o);
      int col = h * HD + dt * 16 + m;
      int r0  = mt * 16 + lg;
#pragma unroll
      for (int v = 0; v < 8; ++v) B2[(r0 + v) * XS + col] = (bf16)o[v];  // O row-major
    }
  }
  __syncthreads();

  // ---- output projection: O(B2) @ Wp + bp -> global f32 ----
  {
    const bf16* Wp    = wpack + 3 * WELEMS;
    float*      og    = out + (size_t)b * NTOK * DIMC;
    const int   mt    = wave >> 1;
    const int   nbase = (wave & 1) * 6;
    bf16x16 A[6];
#pragma unroll
    for (int kt = 0; kt < 6; ++kt) A[kt] = ldsA(B2, mt * 16, kt * 32, XS, lane);
    f32x8 acc[6];
#pragma unroll
    for (int i = 0; i < 6; ++i) acc[i] = (f32x8){0.f,0.f,0.f,0.f,0.f,0.f,0.f,0.f};
#pragma unroll
    for (int kt = 0; kt < 6; ++kt) {
#pragma unroll
      for (int i = 0; i < 6; ++i) {
        bf16x16 bb = wfrag(Wp, (nbase + i) * 6 + kt, lane);
        acc[i] = wmma_bf16(A[kt], bb, acc[i]);
      }
    }
    const int r0 = mt * 16 + ((lane >> 4) << 3);
#pragma unroll
    for (int i = 0; i < 6; ++i) {
      int   col = (nbase + i) * 16 + (lane & 15);
      float bsv = bp[col];
#pragma unroll
      for (int v = 0; v < 8; ++v)
        __builtin_nontemporal_store(acc[i][v] + bsv, og + (r0 + v) * DIMC + col);
    }
  }
}

extern "C" void kernel_launch(void* const* d_in, const int* in_sizes, int n_in,
                              void* d_out, int out_size, void* d_ws, size_t ws_size,
                              hipStream_t stream) {
  const float* xq   = (const float*)d_in[0];
  const float* xk   = (const float*)d_in[1];
  const float* xv   = (const float*)d_in[2];
  const float* mask = (const float*)d_in[3];
  const float* Wq   = (const float*)d_in[4];
  const float* bq   = (const float*)d_in[5];
  const float* Wk   = (const float*)d_in[6];
  const float* bk   = (const float*)d_in[7];
  const float* Wv   = (const float*)d_in[8];
  const float* bv   = (const float*)d_in[9];
  const float* Wp   = (const float*)d_in[10];
  const float* bp   = (const float*)d_in[11];
  const float* bt   = (const float*)d_in[12];
  bf16* wpack = (bf16*)d_ws;   // 4 * 72 * 512 bf16 = 288 KB

  const int Bn = in_sizes[0] / (NTOK * DIMC);  // 4096 windows

  pack_weights<<<dim3((4 * WFRAGS * 32 + 255) / 256), dim3(256), 0, stream>>>(Wq, Wk, Wv, Wp, wpack);
  window_attn<<<dim3(Bn), dim3(256), 0, stream>>>(xq, xk, xv, mask, bq, bk, bv, bp, bt, wpack,
                                                  (float*)d_out);
}
